// Tacotron2_12077448036619
// MI455X (gfx1250) — matrix operand order repaired and checked by reading, not compiled, not run on hardware
//
#include <hip/hip_runtime.h>
#include <hip/hip_bf16.h>
#include <math.h>

// ---------------------------------------------------------------------------
// Tacotron2 forward for MI455X (gfx1250). All dense matmuls (including the
// encoder/postnet convolutions via im2col implicit GEMM) run on
// v_wmma_f32_16x16x32_f16 with fp32 accumulate. Fragment loads are b128
// vector loads of the contiguous per-lane runs from the ISA layouts; M-edge
// checks removed by 16-row padding of activation buffers. Encoder BiLSTM is
// a persistent one-workgroup kernel with LDS-resident state.
// ---------------------------------------------------------------------------

typedef __attribute__((ext_vector_type(16))) _Float16 v16h;
typedef __attribute__((ext_vector_type(8)))  float    v8f;
typedef __attribute__((ext_vector_type(4)))  float    v4f;

#define B_SZ   8
#define TTEXT  192
#define TMEL   200
#define EMB    512
#define NMEL   80

__device__ __forceinline__ float rcp_(float x) { return __builtin_amdgcn_rcpf(x); }
__device__ __forceinline__ float sigmoidf_(float x) { return rcp_(1.0f + __expf(-x)); }
__device__ __forceinline__ float tanhf_(float x) {
  x = fminf(fmaxf(x, -15.0f), 15.0f);
  float t = __expf(2.0f * x);
  return (t - 1.0f) * rcp_(t + 1.0f);
}

template<int ACT> __device__ __forceinline__ float act_(float x) {
  if (ACT == 1) return x > 0.0f ? x : 0.0f;   // relu
  if (ACT == 2) return tanhf_(x);             // tanh
  return x;
}

__device__ __forceinline__ v8f wmma_f16(v16h a, v16h b, v8f c) {
  return __builtin_amdgcn_wmma_f32_16x16x32_f16(false, a, false, b, (short)0, c, false, false);
}

// ---------------------------------------------------------------------------
// Fragment loaders (see ISA 16-bit A 16x32 and B 32x16 layouts).
// A: lane l (m=l&15, h=l>>4): runs [kBase+8h,+8) and [kBase+16+8h,+8).
// B: lane l (n=l&15, h=l>>4): runs [kBase+8h,+8) and [kBase+16+8h,+8).
// ---------------------------------------------------------------------------
__device__ __forceinline__ v16h pack16(v4f a0, v4f a1, v4f b0, v4f b1) {
  v16h o;
#pragma unroll
  for (int i = 0; i < 4; ++i) {
    o[i]      = (_Float16)a0[i];
    o[4 + i]  = (_Float16)a1[i];
    o[8 + i]  = (_Float16)b0[i];
    o[12 + i] = (_Float16)b1[i];
  }
  return o;
}

__device__ __forceinline__ v16h load_frag_a_fast(const float* A, int lda, int row,
                                                 int kBase, int lane) {
  int r = lane & 15, h = lane >> 4;
  const float* ap = A + (size_t)(row + r) * lda + kBase + (h << 3);
  v4f a0 = *(const v4f*)(ap);
  v4f a1 = *(const v4f*)(ap + 4);
  v4f b0 = *(const v4f*)(ap + 16);
  v4f b1 = *(const v4f*)(ap + 20);
  return pack16(a0, a1, b0, b1);
}

__device__ __forceinline__ v16h load_frag_a_edge(const float* A, int lda, int row,
                                                 int kBase, int K, int lane) {
  int r = lane & 15, h = lane >> 4;
  const float* ap = A + (size_t)(row + r) * lda;
  v16h o;
#pragma unroll
  for (int e = 0; e < 16; ++e) {
    int k = kBase + (h << 3) + e + ((e >= 8) ? 8 : 0);
    o[e] = (k < K) ? (_Float16)ap[k] : (_Float16)0.0f;
  }
  return o;
}

__device__ __forceinline__ v16h load_frag_b_fast(const float* W, int ldw, int col,
                                                 int kBase, int lane) {
  int r = lane & 15, h = lane >> 4;
  const float* wp = W + (size_t)(col + r) * ldw + kBase + (h << 3);
  v4f a0 = *(const v4f*)(wp);
  v4f a1 = *(const v4f*)(wp + 4);
  v4f b0 = *(const v4f*)(wp + 16);
  v4f b1 = *(const v4f*)(wp + 20);
  return pack16(a0, a1, b0, b1);
}

__device__ __forceinline__ v16h load_frag_b_edge(const float* W, int ldw, int col,
                                                 int N, int kBase, int K, int lane) {
  int r = lane & 15, h = lane >> 4;
  const float* wp = W + (size_t)(col + r) * ldw;
  bool ok = (col + r) < N;
  v16h o;
#pragma unroll
  for (int e = 0; e < 16; ++e) {
    int k = kBase + (h << 3) + e + ((e >= 8) ? 8 : 0);
    o[e] = (ok && k < K) ? (_Float16)wp[k] : (_Float16)0.0f;
  }
  return o;
}

// Shared accumulate loop for one 16x16 tile
__device__ __forceinline__ v8f wmma_tile(const float* A, const float* W,
                                         int M_pad_ok_tm, int tn, int N, int K,
                                         int lane) {
  int tm = M_pad_ok_tm;
  bool nFull = (tn + 16) <= N;
  v8f acc = {};
  for (int kb = 0; kb < K; kb += 32) {
    bool kFull = (kb + 32) <= K;
    v16h a = kFull ? load_frag_a_fast(A, K, tm, kb, lane)
                   : load_frag_a_edge(A, K, tm, kb, K, lane);
    v16h b = (kFull && nFull) ? load_frag_b_fast(W, K, tn, kb, lane)
                              : load_frag_b_edge(W, K, tn, N, kb, K, lane);
    acc = wmma_f16(a, b, acc);
  }
  return acc;
}

// ---------------------------------------------------------------------------
// WMMA GEMM: out(M,N) = act(A(M,K) @ W(N,K)^T + bias [+ out if beta])
// ---------------------------------------------------------------------------
template<int ACT>
__global__ void k_gemm(const float* __restrict__ A, const float* __restrict__ W,
                       const float* __restrict__ bias, float* __restrict__ out,
                       int M, int N, int K, int beta) {
  int tN = (N + 15) >> 4;
  int tM = (M + 15) >> 4;
  int wave = blockIdx.x * (blockDim.x >> 5) + (threadIdx.x >> 5);
  if (wave >= tM * tN) return;
  int tm = (wave / tN) << 4;
  int tn = (wave % tN) << 4;
  int lane = threadIdx.x & 31;
  v8f acc = wmma_tile(A, W, tm, tn, N, K, lane);
  int r = lane & 15, h = lane >> 4;
  int n = tn + r;
  float bi = (bias && n < N) ? bias[n] : 0.0f;
#pragma unroll
  for (int v = 0; v < 8; ++v) {
    int m = tm + v + (h << 3);
    if (m < M && n < N) {
      float x = acc[v] + bi;
      if (beta) x += out[(size_t)m * N + n];
      out[(size_t)m * N + n] = act_<ACT>(x);
    }
  }
}

// ---------------------------------------------------------------------------
// WMMA GEMM with fused conv-bias + eval BatchNorm + activation epilogue.
// Used for the im2col implicit-GEMM convolutions.
// ---------------------------------------------------------------------------
template<int ACT>
__global__ void k_gemm_bn(const float* __restrict__ A, const float* __restrict__ W,
                          const float* __restrict__ cb, const float* __restrict__ ga,
                          const float* __restrict__ be, const float* __restrict__ me,
                          const float* __restrict__ va, float* __restrict__ out,
                          int M, int N, int K) {
  int tN = (N + 15) >> 4;
  int tM = (M + 15) >> 4;
  int wave = blockIdx.x * (blockDim.x >> 5) + (threadIdx.x >> 5);
  if (wave >= tM * tN) return;
  int tm = (wave / tN) << 4;
  int tn = (wave % tN) << 4;
  int lane = threadIdx.x & 31;
  v8f acc = wmma_tile(A, W, tm, tn, N, K, lane);
  int r = lane & 15, h = lane >> 4;
  int n = tn + r;
  float scale = 0.0f, shift = 0.0f;
  if (n < N) {
    scale = ga[n] * rsqrtf(va[n] + 1e-5f);
    shift = (cb[n] - me[n]) * scale + be[n];
  }
#pragma unroll
  for (int v = 0; v < 8; ++v) {
    int m = tm + v + (h << 3);
    if (m < M && n < N)
      out[(size_t)m * N + n] = act_<ACT>(acc[v] * scale + shift);
  }
}

// ---------------------------------------------------------------------------
// im2col for conv1d K=5 pad=2, row-major activations (B*T, C) -> (B*T, C*5)
// Column index (ci*5 + k) matches torch weight (Cout, Cin, 5) flattening.
// ---------------------------------------------------------------------------
__global__ void k_im2col(const float* __restrict__ x, float* __restrict__ col,
                         int Bn, int T, int C, int Kw, int pad) {
  int row = blockIdx.x;            // b*T + t
  int b = row / T, t = row % T;
  int KC = C * Kw;
  for (int i = threadIdx.x; i < KC; i += blockDim.x) {
    int ci = i / Kw, k = i % Kw;
    int tt = t + k - pad;
    float v = 0.0f;
    if (tt >= 0 && tt < T) v = x[((size_t)b * T + tt) * C + ci];
    col[(size_t)row * KC + i] = v;
  }
}

// ---------------------------------------------------------------------------
// Persistent one-workgroup LSTM scan (encoder BiLSTM, batch 8, hidden 256).
// sH padded to 16 rows (rows 8..15 stay zero) -> unconditional ds_load_b128.
// ---------------------------------------------------------------------------
__global__ void k_lstm_seq(const float* __restrict__ gx,   // (T*8, 1024)
                           const float* __restrict__ whh,  // (1024, 256)
                           const float* __restrict__ bhh,  // (1024)
                           float* __restrict__ memory,     // (B, T, 512)
                           int T, int reverse, int dirOffset) {
  __shared__ float sH[16 * 256];
  __shared__ float sC[8 * 256];
  __shared__ float sG[8 * 1024];
  const int tid = threadIdx.x;   // 512 threads = 16 waves
  for (int i = tid; i < 4096; i += 512) sH[i] = 0.0f;
  for (int i = tid; i < 2048; i += 512) sC[i] = 0.0f;
  __syncthreads();
  const int wave = tid >> 5, lane = tid & 31;
  const int r = lane & 15, hh = lane >> 4;
  for (int step = 0; step < T; ++step) {
    int t = reverse ? (T - 1 - step) : step;
    if (step + 1 < T) {
      int tn2 = reverse ? (T - 2 - step) : (step + 1);
      __builtin_prefetch(gx + ((size_t)tn2 * 8) * 1024 + tid * 16, 0, 1);
    }
    for (int tile = wave; tile < 64; tile += 16) {
      int tn = tile << 4;
      v8f acc = {};
      for (int kb = 0; kb < 256; kb += 32) {
        v16h a = load_frag_a_fast(sH, 256, 0, kb, lane);
        v16h b = load_frag_b_fast(whh, 256, tn, kb, lane);
        acc = wmma_f16(a, b, acc);
      }
#pragma unroll
      for (int v = 0; v < 8; ++v) {
        int m = v + (hh << 3);
        int n = tn + r;
        if (m < 8)
          sG[m * 1024 + n] = acc[v] + bhh[n] + gx[((size_t)t * 8 + m) * 1024 + n];
      }
    }
    __syncthreads();
    for (int i = tid; i < 2048; i += 512) {
      int b = i >> 8, j = i & 255;
      const float* g = sG + b * 1024;
      float ig = sigmoidf_(g[j]);
      float fg = sigmoidf_(g[256 + j]);
      float gg = tanhf_(g[512 + j]);
      float og = sigmoidf_(g[768 + j]);
      float cn = fg * sC[i] + ig * gg;
      float hn = og * tanhf_(cn);
      sC[i] = cn;
      sH[i] = hn;
      memory[((size_t)b * T + t) * 512 + dirOffset + j] = hn;
    }
    __syncthreads();
  }
}

// ---------------------------------------------------------------------------
// Elementwise LSTM cell (decoder LSTMs, hidden 1024)
// ---------------------------------------------------------------------------
__global__ void k_lstm_cell(const float* __restrict__ gates, float* __restrict__ h,
                            float* __restrict__ c, int Bn, int H) {
  int i = blockIdx.x * blockDim.x + threadIdx.x;
  if (i >= Bn * H) return;
  int b = i / H, j = i % H;
  const float* g = gates + (size_t)b * 4 * H;
  float ig = sigmoidf_(g[j]);
  float fg = sigmoidf_(g[H + j]);
  float gg = tanhf_(g[2 * H + j]);
  float og = sigmoidf_(g[3 * H + j]);
  float cn = fg * c[i] + ig * gg;
  c[i] = cn;
  h[i] = og * tanhf_(cn);
}

// ---------------------------------------------------------------------------
// Fused location-sensitive attention, one block per batch element.
// ---------------------------------------------------------------------------
__global__ void k_attention(const float* __restrict__ pq,     // (8,128)
                            const float* __restrict__ pm,     // (8,T,128)
                            const float* __restrict__ memory, // (8,T,512)
                            const float* __restrict__ lw,     // (32,2,31)
                            const float* __restrict__ lb,     // (32)
                            const float* __restrict__ ld,     // (128,32)
                            const float* __restrict__ vatt,   // (128)
                            float* __restrict__ aw, float* __restrict__ awc,
                            float* __restrict__ ctx, int Tt) {
  const int b = blockIdx.x;
  const int tid = threadIdx.x;  // 256
  __shared__ float sE[TTEXT];
  __shared__ float sA[TTEXT];
  __shared__ float sPq[128];
  __shared__ float sRed[256];
  if (tid < 128) sPq[tid] = pq[b * 128 + tid];
  __syncthreads();
  for (int t = tid; t < Tt; t += 256) {
    float loc[32];
#pragma unroll
    for (int ch = 0; ch < 32; ++ch) loc[ch] = lb[ch];
    for (int k = 0; k < 31; ++k) {
      int pos = t + k - 15;
      if (pos < 0 || pos >= Tt) continue;
      float x0 = aw[b * Tt + pos];
      float x1 = awc[b * Tt + pos];
#pragma unroll
      for (int ch = 0; ch < 32; ++ch)
        loc[ch] += lw[(ch * 2 + 0) * 31 + k] * x0 + lw[(ch * 2 + 1) * 31 + k] * x1;
    }
    float e = 0.0f;
    for (int d = 0; d < 128; ++d) {
      float s = sPq[d] + pm[((size_t)b * Tt + t) * 128 + d];
#pragma unroll
      for (int ch = 0; ch < 32; ++ch) s += loc[ch] * ld[d * 32 + ch];
      e += tanhf_(s) * vatt[d];
    }
    sE[t] = e;
  }
  __syncthreads();
  float lm = -1e30f;
  for (int t = tid; t < Tt; t += 256) lm = fmaxf(lm, sE[t]);
  sRed[tid] = lm; __syncthreads();
  for (int s = 128; s > 0; s >>= 1) {
    if (tid < s) sRed[tid] = fmaxf(sRed[tid], sRed[tid + s]);
    __syncthreads();
  }
  float mx = sRed[0]; __syncthreads();
  float ls = 0.0f;
  for (int t = tid; t < Tt; t += 256) ls += __expf(sE[t] - mx);
  sRed[tid] = ls; __syncthreads();
  for (int s = 128; s > 0; s >>= 1) {
    if (tid < s) sRed[tid] += sRed[tid + s];
    __syncthreads();
  }
  float inv = rcp_(sRed[0]); __syncthreads();
  for (int t = tid; t < Tt; t += 256) {
    float a = __expf(sE[t] - mx) * inv;
    sA[t] = a;
    aw[b * Tt + t] = a;
    awc[b * Tt + t] += a;
  }
  __syncthreads();
  for (int d = tid; d < 512; d += 256) {
    float acc = 0.0f;
    for (int t = 0; t < Tt; ++t) acc += sA[t] * memory[((size_t)b * Tt + t) * 512 + d];
    ctx[b * 512 + d] = acc;
  }
}

// ---------------------------------------------------------------------------
// Data movement kernels
// ---------------------------------------------------------------------------
__global__ void k_embed(const int* __restrict__ tok, const float* __restrict__ emb,
                        float* __restrict__ x, int Bn, int T, int C) {
  int row = blockIdx.x;            // b*T + t, output row-major (B*T, C)
  int b = row / T, t = row % T;
  int s = tok[b * T + t];
  for (int c = threadIdx.x; c < C; c += blockDim.x)
    x[(size_t)row * C + c] = emb[(size_t)s * C + c];
}

// (B*T, C) rows (b,t)  ->  (T*B, C) rows (t,b)
__global__ void k_btc_to_tbc(const float* __restrict__ x, float* __restrict__ y,
                             int Bn, int C, int T) {
  int row = blockIdx.x;            // t*Bn + b
  int t = row / Bn, b = row % Bn;
  for (int c = threadIdx.x; c < C; c += blockDim.x)
    y[(size_t)row * C + c] = x[((size_t)b * T + t) * C + c];
}

// (T*B, C) rows (t,b)  ->  (B*T, C) rows (b,t)
__global__ void k_tbc_to_btc(const float* __restrict__ x, float* __restrict__ y,
                             int Bn, int C, int T) {
  int row = blockIdx.x;            // b*T + t
  int b = row / T, t = row % T;
  for (int c = threadIdx.x; c < C; c += blockDim.x)
    y[(size_t)row * C + c] = x[((size_t)t * Bn + b) * C + c];
}

__global__ void k_prenet_in(const float* __restrict__ tm, float* __restrict__ y,
                            int Bn, int C, int T) {
  int tb = blockIdx.x;
  int t = tb / Bn, b = tb % Bn;
  for (int c = threadIdx.x; c < C; c += blockDim.x)
    y[((size_t)t * Bn + b) * C + c] = (t == 0) ? 0.0f : tm[((size_t)b * C + c) * T + (t - 1)];
}

__global__ void k_concat2(float* __restrict__ dst, const float* __restrict__ a, int na,
                          const float* __restrict__ b2, int nb, int rows) {
  int i = blockIdx.x * blockDim.x + threadIdx.x;
  int w = na + nb;
  if (i >= rows * w) return;
  int rr = i / w, col = i % w;
  dst[i] = (col < na) ? a[rr * na + col] : b2[rr * nb + (col - na)];
}

__global__ void k_zero(float* __restrict__ p, int n) {
  int i = blockIdx.x * blockDim.x + threadIdx.x;
  if (i < n) p[i] = 0.0f;
}

// d_out (B,80,T) = melrow (B*T,80) + postrow (B*T,80)
__global__ void k_out(const float* __restrict__ mel, const float* __restrict__ post,
                      float* __restrict__ out, int Bn, int C, int T) {
  int bc = blockIdx.x;
  int b = bc / C, c = bc % C;
  for (int t = threadIdx.x; t < T; t += blockDim.x)
    out[((size_t)b * C + c) * T + t] =
        mel[((size_t)b * T + t) * C + c] + post[((size_t)b * T + t) * C + c];
}

// ---------------------------------------------------------------------------
// Host orchestration
// ---------------------------------------------------------------------------
extern "C" void kernel_launch(void* const* d_in, const int* in_sizes, int n_in,
                              void* d_out, int out_size, void* d_ws, size_t ws_size,
                              hipStream_t stream) {
  (void)in_sizes; (void)n_in; (void)out_size; (void)ws_size;

  // ---- input pointer map (setup_inputs dict order) ----
  const float* emb   = (const float*)d_in[0];
  // enc conv blocks: base 1, 7, 13 ; post conv blocks: base 49, 55, 61, 67, 73
  const float* wih_f = (const float*)d_in[19];
  const float* whh_f = (const float*)d_in[20];
  const float* bih_f = (const float*)d_in[21];
  const float* bhh_f = (const float*)d_in[22];
  const float* wih_b = (const float*)d_in[23];
  const float* whh_b = (const float*)d_in[24];
  const float* bih_b = (const float*)d_in[25];
  const float* bhh_b = (const float*)d_in[26];
  const float* pw1 = (const float*)d_in[27];
  const float* pb1 = (const float*)d_in[28];
  const float* pw2 = (const float*)d_in[29];
  const float* pb2 = (const float*)d_in[30];
  const float* att_wq   = (const float*)d_in[31];
  const float* att_wmem = (const float*)d_in[32];
  const float* att_v    = (const float*)d_in[33];
  const float* loc_w = (const float*)d_in[34];
  const float* loc_b = (const float*)d_in[35];
  const float* loc_d = (const float*)d_in[36];
  const float* arnn_wih = (const float*)d_in[37];
  const float* arnn_whh = (const float*)d_in[38];
  const float* arnn_bih = (const float*)d_in[39];
  const float* arnn_bhh = (const float*)d_in[40];
  const float* drnn_wih = (const float*)d_in[41];
  const float* drnn_whh = (const float*)d_in[42];
  const float* drnn_bih = (const float*)d_in[43];
  const float* drnn_bhh = (const float*)d_in[44];
  const float* proj_w = (const float*)d_in[45];
  const float* proj_b = (const float*)d_in[46];
  // gate head (47,48) unused: final output is mel_out + postnet only
  const int*   tokens  = (const int*)d_in[79];
  const float* teacher = (const float*)d_in[80];

  // ---- workspace layout ----
  float* wsf = (float*)d_ws;
  size_t off = 0;
  auto alloc = [&](size_t n) { float* p = wsf + off; off += (n + 63) & ~(size_t)63; return p; };

  float* rowA    = alloc((size_t)B_SZ * TTEXT * EMB);     // (B*T, 512) row-major
  float* rowB    = alloc((size_t)B_SZ * TTEXT * EMB);
  float* col     = alloc((size_t)B_SZ * TMEL * EMB * 5);  // im2col scratch (max 1600x2560)
  float* xs      = alloc((size_t)TTEXT * B_SZ * EMB);     // (T,B,512)
  float* gx_f    = alloc((size_t)TTEXT * B_SZ * 1024);
  float* gx_b    = alloc((size_t)TTEXT * B_SZ * 1024);
  float* memory  = alloc((size_t)B_SZ * TTEXT * 512);     // (B,T,512)
  float* pm      = alloc((size_t)B_SZ * TTEXT * 128);     // (B,T,128)
  float* pre0    = alloc((size_t)TMEL * B_SZ * NMEL);
  float* pre1    = alloc((size_t)TMEL * B_SZ * 256);
  float* pre2    = alloc((size_t)TMEL * B_SZ * 256);
  // decoder state span (zeroed once; 16-row padding stays zero forever)
  float* ah      = alloc(16 * 1024);
  float* ac      = alloc(B_SZ * 1024);
  float* dh      = alloc(16 * 1024);
  float* dc      = alloc(B_SZ * 1024);
  float* ctx     = alloc(B_SZ * 512);
  float* aw      = alloc(B_SZ * TTEXT);
  float* awc     = alloc(B_SZ * TTEXT);
  float* cell_in = alloc(16 * 768);
  float* dec_in2 = alloc(16 * 1536);
  float* proj_in = alloc(16 * 1536);
  size_t state_end = off;
  float* gates   = alloc(B_SZ * 4096);
  float* pq      = alloc(B_SZ * 128);
  float* mels    = alloc((size_t)TMEL * B_SZ * NMEL);     // (T,B,80)
  float* melrow  = alloc((size_t)B_SZ * TMEL * NMEL);     // (B*T,80)
  float* prowA   = alloc((size_t)B_SZ * TMEL * 512);      // postnet rows
  float* prowB   = alloc((size_t)B_SZ * TMEL * 512);
  float* postrow = alloc((size_t)B_SZ * TMEL * NMEL);

  auto gemm = [&](const float* A, const float* W, const float* bias, float* out,
                  int M, int N, int K, int beta, int act) {
    int tiles = ((M + 15) / 16) * ((N + 15) / 16);
    dim3 grid((tiles + 3) / 4), blk(128);
    if (act == 1)      k_gemm<1><<<grid, blk, 0, stream>>>(A, W, bias, out, M, N, K, beta);
    else if (act == 2) k_gemm<2><<<grid, blk, 0, stream>>>(A, W, bias, out, M, N, K, beta);
    else               k_gemm<0><<<grid, blk, 0, stream>>>(A, W, bias, out, M, N, K, beta);
  };
  // implicit-GEMM conv: im2col + WMMA GEMM with fused conv-bias+BN+act
  auto conv = [&](const float* xrow, int base, float* yrow, int Cin, int Cout,
                  int T, int act) {
    int Kc = Cin * 5;
    k_im2col<<<dim3(B_SZ * T), dim3(256), 0, stream>>>(xrow, col, B_SZ, T, Cin, 5, 2);
    int M = B_SZ * T;
    int tiles = ((M + 15) / 16) * ((Cout + 15) / 16);
    dim3 grid((tiles + 3) / 4), blk(128);
    const float* W  = (const float*)d_in[base];
    const float* cb = (const float*)d_in[base + 1];
    const float* ga = (const float*)d_in[base + 2];
    const float* be = (const float*)d_in[base + 3];
    const float* me = (const float*)d_in[base + 4];
    const float* va = (const float*)d_in[base + 5];
    if (act == 1)
      k_gemm_bn<1><<<grid, blk, 0, stream>>>(col, W, cb, ga, be, me, va, yrow, M, Cout, Kc);
    else if (act == 2)
      k_gemm_bn<2><<<grid, blk, 0, stream>>>(col, W, cb, ga, be, me, va, yrow, M, Cout, Kc);
    else
      k_gemm_bn<0><<<grid, blk, 0, stream>>>(col, W, cb, ga, be, me, va, yrow, M, Cout, Kc);
  };

  // ---- encoder: embed -> 3x(conv+BN+relu) as implicit GEMM -> BiLSTM ----
  k_embed<<<B_SZ * TTEXT, 256, 0, stream>>>(tokens, emb, rowA, B_SZ, TTEXT, EMB);
  conv(rowA, 1, rowB, EMB, EMB, TTEXT, 1);
  conv(rowB, 7, rowA, EMB, EMB, TTEXT, 1);
  conv(rowA, 13, rowB, EMB, EMB, TTEXT, 1);
  k_btc_to_tbc<<<TTEXT * B_SZ, 256, 0, stream>>>(rowB, xs, B_SZ, EMB, TTEXT);
  gemm(xs, wih_f, bih_f, gx_f, TTEXT * B_SZ, 1024, EMB, 0, 0);
  gemm(xs, wih_b, bih_b, gx_b, TTEXT * B_SZ, 1024, EMB, 0, 0);
  k_lstm_seq<<<1, 512, 0, stream>>>(gx_f, whh_f, bhh_f, memory, TTEXT, 0, 0);
  k_lstm_seq<<<1, 512, 0, stream>>>(gx_b, whh_b, bhh_b, memory, TTEXT, 1, 256);
  gemm(memory, att_wmem, nullptr, pm, B_SZ * TTEXT, 128, 512, 0, 0);

  // ---- prenet over all timesteps at once ----
  k_prenet_in<<<TMEL * B_SZ, 128, 0, stream>>>(teacher, pre0, B_SZ, NMEL, TMEL);
  gemm(pre0, pw1, pb1, pre1, TMEL * B_SZ, 256, NMEL, 0, 1);
  gemm(pre1, pw2, pb2, pre2, TMEL * B_SZ, 256, 256, 0, 1);

  // ---- decoder scan ----
  {
    int nstate = (int)(state_end - (size_t)(ah - wsf));
    k_zero<<<(nstate + 255) / 256, 256, 0, stream>>>(ah, nstate);
  }
  for (int t = 0; t < TMEL; ++t) {
    k_concat2<<<(8 * 768 + 255) / 256, 256, 0, stream>>>(
        cell_in, pre2 + (size_t)t * 8 * 256, 256, ctx, 512, 8);
    gemm(cell_in, arnn_wih, arnn_bih, gates, 8, 4096, 768, 0, 0);
    gemm(ah, arnn_whh, arnn_bhh, gates, 8, 4096, 1024, 1, 0);
    k_lstm_cell<<<32, 256, 0, stream>>>(gates, ah, ac, 8, 1024);
    gemm(ah, att_wq, nullptr, pq, 8, 128, 1024, 0, 0);
    k_attention<<<8, 256, 0, stream>>>(pq, pm, memory, loc_w, loc_b, loc_d, att_v,
                                       aw, awc, ctx, TTEXT);
    k_concat2<<<(8 * 1536 + 255) / 256, 256, 0, stream>>>(dec_in2, ah, 1024, ctx, 512, 8);
    gemm(dec_in2, drnn_wih, drnn_bih, gates, 8, 4096, 1536, 0, 0);
    gemm(dh, drnn_whh, drnn_bhh, gates, 8, 4096, 1024, 1, 0);
    k_lstm_cell<<<32, 256, 0, stream>>>(gates, dh, dc, 8, 1024);
    k_concat2<<<(8 * 1536 + 255) / 256, 256, 0, stream>>>(proj_in, dh, 1024, ctx, 512, 8);
    gemm(proj_in, proj_w, proj_b, mels + (size_t)t * 8 * NMEL, 8, NMEL, 1536, 0, 0);
  }
  k_tbc_to_btc<<<B_SZ * TMEL, 128, 0, stream>>>(mels, melrow, B_SZ, NMEL, TMEL);

  // ---- postnet (implicit-GEMM convs) + residual ----
  conv(melrow, 49, prowA, NMEL, 512, TMEL, 2);
  conv(prowA, 55, prowB, 512, 512, TMEL, 2);
  conv(prowB, 61, prowA, 512, 512, TMEL, 2);
  conv(prowA, 67, prowB, 512, 512, TMEL, 2);
  conv(prowB, 73, postrow, 512, NMEL, TMEL, 0);
  k_out<<<B_SZ * NMEL, 256, 0, stream>>>(melrow, postrow, (float*)d_out,
                                         B_SZ, NMEL, TMEL);
}
